// NonLocalBlock_44014824849661
// MI455X (gfx1250) — matrix operand order repaired and checked by reading, not compiled, not run on hardware
//
#include <hip/hip_runtime.h>
#include <hip/hip_bf16.h>

#define BATCH 8
#define CH    512
#define ICH   256
#define HW    2304
#define NROW  (BATCH * HW)

typedef __attribute__((ext_vector_type(16))) __bf16 v16bf;
typedef __attribute__((ext_vector_type(8)))  __bf16 v8bf;
typedef __attribute__((ext_vector_type(8)))  float  v8f;

// ---------------------------------------------------------------------------
// WMMA helpers (CDNA5 wave32, D = A(16x32 bf16) * B(32x16 bf16) + C(16x16 f32))
// ---------------------------------------------------------------------------
__device__ __forceinline__ v8f zero_v8f() {
  v8f z;
#pragma unroll
  for (int i = 0; i < 8; ++i) z[i] = 0.0f;
  return z;
}

// A-matrix 16x32 (MxK), row-major source with leading dim `ld`.
// ISA layout: lanes 0-15 -> M=lane, K={0..7, 16..23}; lanes 16-31 -> M=lane-16, K={8..15, 24..31}
__device__ __forceinline__ v16bf load_a_frag(const __bf16* base, int ld, int lane) {
  const int half = lane >> 4, lr = lane & 15;
  const __bf16* row = base + (size_t)lr * ld + half * 8;
  v8bf lo = *(const v8bf*)(row);
  v8bf hi = *(const v8bf*)(row + 16);
  v16bf a;
#pragma unroll
  for (int i = 0; i < 8; ++i) { a[i] = lo[i]; a[i + 8] = hi[i]; }
  return a;
}

// B-matrix 32x16 (KxN). Source stored N-major (i.e. B^T row-major, like weights O x K):
// lane (0-15): col N=lane, K={0..7, 16..23}; lane (16-31): col N=lane-16, K={8..15, 24..31}.
__device__ __forceinline__ v16bf load_b_frag(const __bf16* base, int ld, int lane) {
  const int half = lane >> 4, lr = lane & 15;
  const __bf16* p = base + (size_t)lr * ld + half * 8;
  v8bf lo = *(const v8bf*)(p);
  v8bf hi = *(const v8bf*)(p + 16);
  v16bf b;
#pragma unroll
  for (int i = 0; i < 8; ++i) { b[i] = lo[i]; b[i + 8] = hi[i]; }
  return b;
}

__device__ __forceinline__ v8f wmma_bf16(v16bf a, v16bf b, v8f c) {
  return __builtin_amdgcn_wmma_f32_16x16x32_bf16(false, a, false, b, (short)0, c,
                                                 false, false);
}

// ---------------------------------------------------------------------------
// K1a: transpose+convert x (b,512,hw) f32 -> xT (b*hw, 512) bf16
// ---------------------------------------------------------------------------
__global__ __launch_bounds__(256) void pack_x(const float* __restrict__ x,
                                              __bf16* __restrict__ xT) {
  __shared__ float t[32][33];
  const int tx = threadIdx.x, ty = threadIdx.y;
  const int p0 = blockIdx.x * 32, c0 = blockIdx.y * 32, b = blockIdx.z;
#pragma unroll
  for (int k = 0; k < 4; ++k)
    t[ty + k * 8][tx] = x[((size_t)b * CH + c0 + ty + k * 8) * HW + p0 + tx];
  __syncthreads();
#pragma unroll
  for (int k = 0; k < 4; ++k)
    xT[((size_t)b * HW + p0 + ty + k * 8) * CH + c0 + tx] = (__bf16)t[tx][ty + k * 8];
}

// ---------------------------------------------------------------------------
// K1b: convert weights to bf16; wcat = [theta_w ; g_w ; phi_w] (768 x 512)
// ---------------------------------------------------------------------------
__global__ __launch_bounds__(256) void pack_w(const float* __restrict__ theta_w,
                                              const float* __restrict__ g_w,
                                              const float* __restrict__ phi_w,
                                              const float* __restrict__ w_w,
                                              __bf16* __restrict__ wcat,
                                              __bf16* __restrict__ wwB) {
  const int tid = blockIdx.x * 256 + threadIdx.x;
  if (tid < 768 * 512) {
    const int o = tid >> 9, k = tid & 511;
    float v = (o < 256) ? theta_w[o * 512 + k]
            : (o < 512) ? g_w[(o - 256) * 512 + k]
                        : phi_w[(o - 512) * 512 + k];
    wcat[tid] = (__bf16)v;
  } else {
    const int t2 = tid - 768 * 512;   // < 512*256
    wwB[t2] = (__bf16)w_w[t2];
  }
}

// ---------------------------------------------------------------------------
// K2: projection GEMM (18432 x 512) @ (512 x 768) -> theta/(gT transposed)/phi
// one wave = 16x64 tile, K loop 16 steps of 32
// ---------------------------------------------------------------------------
__global__ __launch_bounds__(256) void gemm_proj(const __bf16* __restrict__ xT,
                                                 const __bf16* __restrict__ wcat,
                                                 const float* __restrict__ theta_b,
                                                 const float* __restrict__ g_b,
                                                 const float* __restrict__ phi_b,
                                                 __bf16* __restrict__ theta,
                                                 __bf16* __restrict__ gT,
                                                 __bf16* __restrict__ phi) {
  const int lane = threadIdx.x & 31;
  const int wave = threadIdx.x >> 5;
  const int gt   = blockIdx.x * 8 + wave;      // 13824 wave-tiles
  const int rowb = gt / 12, colb = gt % 12;    // 1152 row-blocks x 12 col-blocks
  const int half = lane >> 4, lr = lane & 15;
  const __bf16* Ab = xT + (size_t)rowb * 16 * 512;

  v8f acc[4];
#pragma unroll
  for (int j = 0; j < 4; ++j) acc[j] = zero_v8f();

  for (int kc = 0; kc < 16; ++kc) {
    v16bf a = load_a_frag(Ab + kc * 32, 512, lane);
#pragma unroll
    for (int j = 0; j < 4; ++j) {
      v16bf b = load_b_frag(wcat + (size_t)(colb * 64 + j * 16) * 512 + kc * 32, 512, lane);
      acc[j] = wmma_bf16(a, b, acc[j]);
    }
  }
#pragma unroll
  for (int j = 0; j < 4; ++j) {
    const int col = colb * 64 + j * 16 + lr;
    const float bias = (col < 256) ? theta_b[col]
                     : (col < 512) ? g_b[col - 256]
                                   : phi_b[col - 512];
#pragma unroll
    for (int i = 0; i < 8; ++i) {
      const int row = rowb * 16 + i + half * 8;    // D layout: M = i (+8 for hi half)
      const int bb = row / HW, p = row - bb * HW;
      const float v = acc[j][i] + bias;
      if (col < 256)      theta[(size_t)row * 256 + col] = (__bf16)v;
      else if (col < 512) gT[((size_t)bb * 256 + (col - 256)) * HW + p] = (__bf16)v;
      else                phi[(size_t)row * 256 + (col - 512)] = (__bf16)v;
    }
  }
}

// ---------------------------------------------------------------------------
// K3: fused flash attention. One wave owns 16 queries; streams keys 32 at a time.
// S = theta * phi^T (8 wmma x2 tiles), online softmax, O += P * g (16 wmma).
// P relayout (D->A frag) through per-wave private LDS tile.
// ---------------------------------------------------------------------------
__global__ __launch_bounds__(128) void attn_kernel(const __bf16* __restrict__ theta,
                                                   const __bf16* __restrict__ phi,
                                                   const __bf16* __restrict__ gT,
                                                   __bf16* __restrict__ y) {
  __shared__ __bf16 pbuf[4][16 * 32];
  const int lane = threadIdx.x & 31;
  const int wave = threadIdx.x >> 5;
  const int qt   = blockIdx.x * 4 + wave;       // 1152 query-blocks
  const int b    = qt / 144;
  const int q0   = (qt - b * 144) * 16;
  const int half = lane >> 4, lr = lane & 15;
  const __bf16* thB = theta + ((size_t)b * HW + q0) * 256;
  const __bf16* phB = phi + (size_t)b * HW * 256;
  const __bf16* gB  = gT + (size_t)b * 256 * HW;

  v8f O[16];
#pragma unroll
  for (int j = 0; j < 16; ++j) O[j] = zero_v8f();
  float m[8], l[8];
#pragma unroll
  for (int i = 0; i < 8; ++i) { m[i] = -3.0e38f; l[i] = 0.0f; }

  for (int kt = 0; kt < HW; kt += 32) {
    v8f S0 = zero_v8f(), S1 = zero_v8f();
#pragma unroll
    for (int cc = 0; cc < 8; ++cc) {
      v16bf a  = load_a_frag(thB + cc * 32, 256, lane);
      v16bf b0 = load_b_frag(phB + (size_t)kt * 256 + cc * 32, 256, lane);
      v16bf b1 = load_b_frag(phB + (size_t)(kt + 16) * 256 + cc * 32, 256, lane);
      S0 = wmma_bf16(a, b0, S0);
      S1 = wmma_bf16(a, b1, S1);
    }
    // online softmax: rows live in VGPR index i (lanes<16: row i; lanes>=16: row i+8);
    // reduce across the 16 lanes holding the row (xor masks < 16 stay in-half).
    float alpha[8];
#pragma unroll
    for (int i = 0; i < 8; ++i) {
      float mx = fmaxf(S0[i], S1[i]);
#pragma unroll
      for (int d = 1; d < 16; d <<= 1) mx = fmaxf(mx, __shfl_xor(mx, d, 32));
      const float mn = fmaxf(m[i], mx);
      alpha[i] = __expf(m[i] - mn);
      const float p0 = __expf(S0[i] - mn);
      const float p1 = __expf(S1[i] - mn);
      S0[i] = p0; S1[i] = p1;
      float rs = p0 + p1;
#pragma unroll
      for (int d = 1; d < 16; d <<= 1) rs += __shfl_xor(rs, d, 32);
      l[i] = l[i] * alpha[i] + rs;
      m[i] = mn;
    }
#pragma unroll
    for (int j = 0; j < 16; ++j)
#pragma unroll
      for (int i = 0; i < 8; ++i) O[j][i] *= alpha[i];

    // P tile (16q x 32k) D-layout -> row-major bf16 in LDS -> reload as A-frag.
#pragma unroll
    for (int i = 0; i < 8; ++i) {
      const int row = i + half * 8;
      pbuf[wave][row * 32 + lr]      = (__bf16)S0[i];
      pbuf[wave][row * 32 + 16 + lr] = (__bf16)S1[i];
    }
    asm volatile("s_wait_dscnt 0x0" ::: "memory");   // same-wave LDS RAW
    v16bf pf = load_a_frag(&pbuf[wave][0], 32, lane);

#pragma unroll
    for (int j = 0; j < 16; ++j) {
      v16bf gb = load_b_frag(gB + (size_t)(j * 16) * HW + kt, HW, lane);
      O[j] = wmma_bf16(pf, gb, O[j]);
    }
  }

  float inv[8];
#pragma unroll
  for (int i = 0; i < 8; ++i) inv[i] = 1.0f / l[i];
#pragma unroll
  for (int j = 0; j < 16; ++j)
#pragma unroll
    for (int i = 0; i < 8; ++i) {
      const int row = q0 + i + half * 8;
      y[((size_t)b * HW + row) * 256 + j * 16 + lr] = (__bf16)(O[j][i] * inv[i]);
    }
}

// ---------------------------------------------------------------------------
// K4: output GEMM (18432 x 256) @ (256 x 512) + w_b -> wy (row-major N x 512, f32)
// ---------------------------------------------------------------------------
__global__ __launch_bounds__(256) void gemm_out(const __bf16* __restrict__ y,
                                                const __bf16* __restrict__ ww,
                                                const float* __restrict__ wb,
                                                float* __restrict__ wy) {
  const int lane = threadIdx.x & 31;
  const int wave = threadIdx.x >> 5;
  const int gt   = blockIdx.x * 8 + wave;     // 9216 wave-tiles
  const int rowb = gt / 8, colb = gt % 8;
  const int half = lane >> 4, lr = lane & 15;
  const __bf16* Ab = y + (size_t)rowb * 16 * 256;

  v8f acc[4];
#pragma unroll
  for (int j = 0; j < 4; ++j) acc[j] = zero_v8f();

  for (int kc = 0; kc < 8; ++kc) {
    v16bf a = load_a_frag(Ab + kc * 32, 256, lane);
#pragma unroll
    for (int j = 0; j < 4; ++j) {
      v16bf b = load_b_frag(ww + (size_t)(colb * 64 + j * 16) * 256 + kc * 32, 256, lane);
      acc[j] = wmma_bf16(a, b, acc[j]);
    }
  }
#pragma unroll
  for (int j = 0; j < 4; ++j) {
    const int col = colb * 64 + j * 16 + lr;
    const float bias = wb[col];
#pragma unroll
    for (int i = 0; i < 8; ++i) {
      const int row = rowb * 16 + i + half * 8;
      wy[(size_t)row * 512 + col] = acc[j][i] + bias;
    }
  }
}

// ---------------------------------------------------------------------------
// K5a/K5b: deterministic two-pass batch-norm statistics over (b,h,w) per channel
// ---------------------------------------------------------------------------
__global__ __launch_bounds__(256) void bn_stats1(const float* __restrict__ wy,
                                                 float* __restrict__ pS,
                                                 float* __restrict__ pQ) {
  const int blk = blockIdx.x;     // 512 slabs of 36 rows
  const int t = threadIdx.x;      // 256 threads -> 2 channels each
#pragma unroll
  for (int h = 0; h < 2; ++h) {
    const int c = t + h * 256;
    float s = 0.f, q = 0.f;
    for (int r = 0; r < 36; ++r) {
      const float v = wy[((size_t)blk * 36 + r) * 512 + c];
      s += v; q += v * v;
    }
    pS[blk * 512 + c] = s;
    pQ[blk * 512 + c] = q;
  }
}

__global__ __launch_bounds__(512) void bn_stats2(const float* __restrict__ pS,
                                                 const float* __restrict__ pQ,
                                                 const float* __restrict__ gamma,
                                                 const float* __restrict__ beta,
                                                 float* __restrict__ scale,
                                                 float* __restrict__ shift) {
  const int c = threadIdx.x;
  float s = 0.f, q = 0.f;
  for (int i = 0; i < 512; ++i) { s += pS[i * 512 + c]; q += pQ[i * 512 + c]; }
  const float n = (float)NROW;
  const float mean = s / n;
  const float var  = q / n - mean * mean;
  const float sc   = gamma[c] * rsqrtf(var + 1e-5f);
  scale[c] = sc;
  shift[c] = beta[c] - mean * sc;
}

// ---------------------------------------------------------------------------
// K6: out = BN(wy) + x, with LDS tile transpose (wy is (b*hw,512), out is (b,512,hw))
// ---------------------------------------------------------------------------
__global__ __launch_bounds__(256) void bn_final(const float* __restrict__ wy,
                                                const float* __restrict__ x,
                                                const float* __restrict__ scale,
                                                const float* __restrict__ shift,
                                                float* __restrict__ out) {
  __shared__ float t[32][33];
  const int tx = threadIdx.x, ty = threadIdx.y;
  const int p0 = blockIdx.x * 32, c0 = blockIdx.y * 32, b = blockIdx.z;
#pragma unroll
  for (int k = 0; k < 4; ++k)
    t[ty + k * 8][tx] = wy[((size_t)b * HW + p0 + ty + k * 8) * 512 + c0 + tx];
  __syncthreads();
#pragma unroll
  for (int k = 0; k < 4; ++k) {
    const int c = c0 + ty + k * 8;
    const size_t idx = ((size_t)b * CH + c) * HW + p0 + tx;
    out[idx] = t[tx][ty + k * 8] * scale[c] + shift[c] + x[idx];
  }
}

// ---------------------------------------------------------------------------
extern "C" void kernel_launch(void* const* d_in, const int* in_sizes, int n_in,
                              void* d_out, int out_size, void* d_ws, size_t ws_size,
                              hipStream_t stream) {
  (void)in_sizes; (void)n_in; (void)out_size; (void)ws_size;
  const float* x       = (const float*)d_in[0];
  const float* g_w     = (const float*)d_in[1];
  const float* g_b     = (const float*)d_in[2];
  const float* theta_w = (const float*)d_in[3];
  const float* theta_b = (const float*)d_in[4];
  const float* phi_w   = (const float*)d_in[5];
  const float* phi_b   = (const float*)d_in[6];
  const float* w_w     = (const float*)d_in[7];
  const float* w_b     = (const float*)d_in[8];
  const float* gamma   = (const float*)d_in[9];
  const float* beta    = (const float*)d_in[10];
  float* out = (float*)d_out;

  char* ws = (char*)d_ws;
  size_t off = 0;
  auto carve = [&](size_t bytes) -> void* {
    void* p = ws + off;
    off += (bytes + 255) & ~(size_t)255;
    return p;
  };
  __bf16* xT    = (__bf16*)carve((size_t)NROW * 512 * 2);
  __bf16* wcat  = (__bf16*)carve((size_t)768 * 512 * 2);
  __bf16* wwB   = (__bf16*)carve((size_t)512 * 256 * 2);
  __bf16* theta = (__bf16*)carve((size_t)NROW * 256 * 2);
  __bf16* phi   = (__bf16*)carve((size_t)NROW * 256 * 2);
  __bf16* gT    = (__bf16*)carve((size_t)BATCH * 256 * HW * 2);
  __bf16* yb    = (__bf16*)carve((size_t)NROW * 256 * 2);
  float*  wy    = (float*)carve((size_t)NROW * 512 * 4);
  float*  pS    = (float*)carve((size_t)512 * 512 * 4);
  float*  pQ    = (float*)carve((size_t)512 * 512 * 4);
  float*  scale = (float*)carve((size_t)512 * 4);
  float*  shift = (float*)carve((size_t)512 * 4);

  pack_x<<<dim3(HW / 32, CH / 32, BATCH), dim3(32, 8), 0, stream>>>(x, xT);
  pack_w<<<dim3((768 * 512 + 512 * 256) / 256), dim3(256), 0, stream>>>(
      theta_w, g_w, phi_w, w_w, wcat, wwB);
  gemm_proj<<<dim3((NROW / 16) * 12 / 8), dim3(256), 0, stream>>>(
      xT, wcat, theta_b, g_b, phi_b, theta, gT, phi);
  attn_kernel<<<dim3((NROW / 16) / 4), dim3(128), 0, stream>>>(theta, phi, gT, yb);
  gemm_out<<<dim3((NROW / 16) * 8 / 8), dim3(256), 0, stream>>>(yb, wwB, w_b, wy);
  bn_stats1<<<dim3(NROW / 36), dim3(256), 0, stream>>>(wy, pS, pQ);
  bn_stats2<<<dim3(1), dim3(512), 0, stream>>>(pS, pQ, gamma, beta, scale, shift);
  bn_final<<<dim3(HW / 32, CH / 32, BATCH), dim3(32, 8), 0, stream>>>(
      wy, x, scale, shift, out);
}
